// AutoformerEncodeBlock_7851200217098
// MI455X (gfx1250) — matrix operand order repaired and checked by reading, not compiled, not run on hardware
//
#include <hip/hip_runtime.h>
#include <hip/hip_bf16.h>
#include <math.h>

typedef __bf16 bf16_t;
typedef __attribute__((ext_vector_type(16))) __bf16 bf16x16;
typedef __attribute__((ext_vector_type(8)))  __bf16 bf16x8;
typedef __attribute__((ext_vector_type(8)))  float  f32x8;

#define BDIM 256

// ---------------------------------------------------------------------------
// Batched/strided bf16 WMMA GEMM:
//   D[z][m,n] = A[z][m, 0:K] * Bw[z][n, 0:K]^T     (row-major, leading dims lda/ldb/ldd)
// Block: 256 thr = 8 waves (2 x 4); wave tile 64x64 (4x4 wmma); block tile 128x256.
// MODE: 0 = store f32
//       1 = store bf16
//       2 = f32 out = acc + bias[n] + res[m,n]
//       3 = bf16 out = GELU_exact(acc)
//       4 = f32 out = acc + res[m,n]
// ---------------------------------------------------------------------------
template<int MODE>
__global__ __launch_bounds__(BDIM)
void gemm_wmma_bf16(const bf16_t* __restrict__ A,  int lda, size_t strideA,
                    const bf16_t* __restrict__ Bw, int ldb, size_t strideB,
                    float*  __restrict__ Df,
                    bf16_t* __restrict__ Db,       int ldd, size_t strideD,
                    const float* __restrict__ bias,
                    const float* __restrict__ res, int ldres,
                    int K)
{
  const int z = blockIdx.z;
  A  += (size_t)z * strideA;
  Bw += (size_t)z * strideB;

  const int tid  = threadIdx.x;
  const int lane = tid & 31;
  const int wave = tid >> 5;
  const int wm   = wave >> 2;          // 0..1
  const int wn   = wave & 3;           // 0..3
  const int bm   = blockIdx.y * 128 + wm * 64;
  const int bn   = blockIdx.x * 256 + wn * 64;
  const int half = lane >> 4;          // lane group 0 / 1
  const int l16  = lane & 15;

  // A-frag: lanes 0-15 hold K {0..15}; lanes 16-31 hold K {16..31}
  const int amBase = bm + l16;
  const int koffA  = half * 16;
  // B-frag: lane holds W-row n, 16 contiguous K at offset half*16
  const int bnBase = bn + l16;
  const int koffB  = half * 16;

  f32x8 acc[4][4] = {};

  for (int kk = 0; kk < K; kk += 32) {
    bf16x16 a[4], b[4];
#pragma unroll
    for (int tm = 0; tm < 4; ++tm) {
      const bf16_t* p = A + (size_t)(amBase + tm * 16) * lda + kk + koffA;
      bf16x8 lo = *(const bf16x8*)(p);
      bf16x8 hi = *(const bf16x8*)(p + 8);
#pragma unroll
      for (int j = 0; j < 8; ++j) { a[tm][j] = lo[j]; a[tm][8 + j] = hi[j]; }
    }
#pragma unroll
    for (int tn = 0; tn < 4; ++tn) {
      b[tn] = *(const bf16x16*)(Bw + (size_t)(bnBase + tn * 16) * ldb + kk + koffB);
    }
    if (kk + 96 < K) {   // prefetch 2 k-steps ahead (global_prefetch_b8)
      __builtin_prefetch(A  + (size_t)amBase * lda + kk + 64 + koffA, 0, 3);
      __builtin_prefetch(Bw + (size_t)bnBase * ldb + kk + 64 + koffB, 0, 3);
    }
#pragma unroll
    for (int tm = 0; tm < 4; ++tm)
#pragma unroll
      for (int tn = 0; tn < 4; ++tn)
        acc[tm][tn] = __builtin_amdgcn_wmma_f32_16x16x32_bf16(
            false, a[tm], false, b[tn], (short)0, acc[tm][tn], false, false);
  }

#pragma unroll
  for (int tm = 0; tm < 4; ++tm) {
#pragma unroll
    for (int tn = 0; tn < 4; ++tn) {
      const int row0 = bm + tm * 16 + half * 8;
      const int col  = bn + tn * 16 + l16;
#pragma unroll
      for (int r = 0; r < 8; ++r) {
        const int m = row0 + r;
        const size_t off = (size_t)z * strideD + (size_t)m * ldd + col;
        float v = acc[tm][tn][r];
        if (MODE == 0)      { Df[off] = v; }
        else if (MODE == 1) { Db[off] = (bf16_t)v; }
        else if (MODE == 2) { Df[off] = v + bias[col] + res[(size_t)m * ldres + col]; }
        else if (MODE == 3) { float g = 0.5f * v * (1.0f + erff(v * 0.70710678118654752f));
                              Db[off] = (bf16_t)g; }
        else                { Df[off] = v + res[(size_t)m * ldres + col]; }
      }
    }
  }
}

// ---------------------------------------------------------------------------
// LayerNorm over last dim (C), output bf16.  One block per row.
// ---------------------------------------------------------------------------
__global__ __launch_bounds__(BDIM)
void layernorm_bf16(const float* __restrict__ x, const float* __restrict__ g,
                    const float* __restrict__ bta, bf16_t* __restrict__ out, int C)
{
  const int row = blockIdx.x;
  const float* xr = x + (size_t)row * C;
  float s = 0.f, s2 = 0.f;
  for (int c = threadIdx.x; c < C; c += BDIM) { float v = xr[c]; s += v; s2 += v * v; }
  __shared__ float r0[BDIM], r1[BDIM];
  r0[threadIdx.x] = s; r1[threadIdx.x] = s2;
  __syncthreads();
  for (int off = BDIM / 2; off > 0; off >>= 1) {
    if (threadIdx.x < off) { r0[threadIdx.x] += r0[threadIdx.x + off];
                             r1[threadIdx.x] += r1[threadIdx.x + off]; }
    __syncthreads();
  }
  const float mu  = r0[0] / C;
  const float var = r1[0] / C - mu * mu;
  const float inv = rsqrtf(var + 1e-5f);
  for (int c = threadIdx.x; c < C; c += BDIM) {
    float v = (xr[c] - mu) * inv * g[c] + bta[c];
    out[(size_t)row * C + c] = (bf16_t)v;
  }
}

__global__ __launch_bounds__(BDIM)
void cvt_f32_to_bf16(const float* __restrict__ in, bf16_t* __restrict__ out, int n)
{
  int i = blockIdx.x * BDIM + threadIdx.x;
  if (i < n) out[i] = (bf16_t)in[i];
}

// ---------------------------------------------------------------------------
// Stage 1 of diagonal means: each block owns a 128-column strip of P[b] and
// scatters elements into per-wave LDS histograms over tau=(r-t)&(L-1).
// Coalesced global reads; per-wave bins => deterministic, race-free.
// grid = (L/128, B), block = 256 (8 waves).  L must be a power of two.
// ---------------------------------------------------------------------------
__global__ __launch_bounds__(BDIM)
void diag_partial(const float* __restrict__ P, float* __restrict__ partial, int L)
{
  const int b  = blockIdx.y;
  const int tc = blockIdx.x;
  const int t0 = tc * 128;
  const int LM = L - 1;
  __shared__ float bins[8][2048];          // 64 KB
  for (int i = threadIdx.x; i < 8 * 2048; i += BDIM) (&bins[0][0])[i] = 0.f;
  __syncthreads();

  const int wave  = threadIdx.x >> 5;
  const int lane  = threadIdx.x & 31;
  const int col   = t0 + (wave & 3) * 32 + lane;   // 32 consecutive cols per wave
  const int rbase = (wave >> 2) * (L / 2);         // waves 0-3: low half, 4-7: high half
  const float* Pb = P + (size_t)b * L * L;
  float* mybins = bins[wave];

#pragma unroll 4
  for (int rr = 0; rr < 2048 / 2; ++rr) {          // L/2 rows per wave
    const int r = rbase + rr;
    const float v = Pb[(size_t)r * L + col];
    const int tau = (r - col + L) & LM;            // lanes hit consecutive bins
    mybins[tau] += v;
  }
  __syncthreads();

  float* out = partial + ((size_t)b * gridDim.x + tc) * L;
  for (int tau = threadIdx.x; tau < L; tau += BDIM) {
    float s = 0.f;
#pragma unroll
    for (int w = 0; w < 8; ++w) s += bins[w][tau];
    out[tau] = s;
  }
}

// Stage 2: mc[b,tau] = scale * sum_chunks partial[b][c][tau].  grid = (L/BDIM, B)
__global__ __launch_bounds__(BDIM)
void diag_finalize(const float* __restrict__ partial, float* __restrict__ mc,
                   int L, int nchunks, float scale)
{
  const int b   = blockIdx.y;
  const int tau = blockIdx.x * BDIM + threadIdx.x;
  float s = 0.f;
  for (int c = 0; c < nchunks; ++c)
    s += partial[((size_t)b * nchunks + c) * L + tau];
  mc[(size_t)b * L + tau] = s * scale;
}

// top-k (k<=8) + softmax over mc[b, 0..L-1].  One block per batch.  L <= 2048.
__global__ __launch_bounds__(BDIM)
void topk_softmax(const float* __restrict__ mc, float* __restrict__ alpha,
                  int* __restrict__ delay, int L, int k)
{
  const int b = blockIdx.x;
  __shared__ float vals[2048];
  __shared__ float rv[BDIM];
  __shared__ int   ri[BDIM];
  __shared__ float wsel[8];
  __shared__ int   isel[8];
  for (int i = threadIdx.x; i < L; i += BDIM) vals[i] = mc[(size_t)b * L + i];
  __syncthreads();
  for (int it = 0; it < k; ++it) {
    float best = -3.4e38f; int bi = 0;
    for (int i = threadIdx.x; i < L; i += BDIM) {
      float v = vals[i];
      if (v > best) { best = v; bi = i; }
    }
    rv[threadIdx.x] = best; ri[threadIdx.x] = bi;
    __syncthreads();
    for (int off = BDIM / 2; off > 0; off >>= 1) {
      if (threadIdx.x < off) {
        if (rv[threadIdx.x + off] > rv[threadIdx.x]) {
          rv[threadIdx.x] = rv[threadIdx.x + off];
          ri[threadIdx.x] = ri[threadIdx.x + off];
        }
      }
      __syncthreads();
    }
    if (threadIdx.x == 0) { wsel[it] = rv[0]; isel[it] = ri[0]; vals[ri[0]] = -3.4e38f; }
    __syncthreads();
  }
  if (threadIdx.x == 0) {
    float mx = wsel[0];
    for (int i = 1; i < k; ++i) mx = fmaxf(mx, wsel[i]);
    float e[8]; float se = 0.f;
    for (int i = 0; i < k; ++i) { e[i] = __expf(wsel[i] - mx); se += e[i]; }
    for (int i = 0; i < k; ++i) { alpha[b * k + i] = e[i] / se; delay[b * k + i] = isel[i]; }
  }
}

// agg[b,l,c] = bf16( sum_i alpha[b,i] * v[b,(l+delay[b,i])%L,c] )
// v is the bf16 slice qkv[:, :, 2048:3072] with row stride ldv.
__global__ __launch_bounds__(BDIM)
void aggregate_topk(const bf16_t* __restrict__ v, int ldv,
                    const float* __restrict__ alpha, const int* __restrict__ delay,
                    bf16_t* __restrict__ aggb, int L, int C, int k)
{
  const int b = blockIdx.y;
  const size_t idx = (size_t)blockIdx.x * BDIM + threadIdx.x;  // within [L*C)
  const int l = (int)(idx / C);
  const int c = (int)(idx % C);
  float s = 0.f;
  for (int i = 0; i < k; ++i) {
    int d = delay[b * k + i];
    int lr = l + d; if (lr >= L) lr -= L;
    s += alpha[b * k + i] * (float)v[((size_t)b * L + lr) * ldv + c];
  }
  aggb[(size_t)b * L * C + idx] = (bf16_t)s;
}

// out = in - moving_avg_25(in) along L with edge-replicate padding
__global__ __launch_bounds__(BDIM)
void movavg_sub(const float* __restrict__ in, float* __restrict__ out, int L, int C)
{
  const int b = blockIdx.y;
  const size_t idx = (size_t)blockIdx.x * BDIM + threadIdx.x;  // within [L*C)
  const int l = (int)(idx / C);
  const int c = (int)(idx % C);
  const float* base = in + (size_t)b * L * C + c;
  float s = 0.f;
#pragma unroll
  for (int j = -12; j <= 12; ++j) {
    int ll = l + j;
    ll = ll < 0 ? 0 : (ll >= L ? L - 1 : ll);
    s += base[(size_t)ll * C];
  }
  out[(size_t)b * L * C + idx] = in[(size_t)b * L * C + idx] - s * (1.0f / 25.0f);
}

// ---------------------------------------------------------------------------
extern "C" void kernel_launch(void* const* d_in, const int* in_sizes, int n_in,
                              void* d_out, int out_size, void* d_ws, size_t ws_size,
                              hipStream_t stream)
{
  (void)in_sizes; (void)n_in; (void)out_size; (void)ws_size;
  const int B = 8, L = 2048, C = 1024, Hd = 4096, TK = 7;
  const int M = B * L;                       // 16384
  const int C3 = 3 * C;                      // fused qkv width
  const int NCH = L / 128;                   // diag col-chunks = 16

  const float* x    = (const float*)d_in[0];
  const float* Wq   = (const float*)d_in[1];
  const float* Wk   = (const float*)d_in[2];
  const float* Wv   = (const float*)d_in[3];
  const float* Wo   = (const float*)d_in[4];
  const float* bo   = (const float*)d_in[5];
  const float* ln_g = (const float*)d_in[6];
  const float* ln_b = (const float*)d_in[7];
  const float* W1   = (const float*)d_in[8];
  const float* W2   = (const float*)d_in[9];

  // ---- workspace layout (aliased across dead lifetimes) ----
  char* ws = (char*)d_ws;
  size_t cur = 0;
  auto take = [&](size_t bytes) { void* p = ws + cur; cur = (cur + bytes + 255) & ~(size_t)255; return p; };

  bf16_t* xnb   = (bf16_t*)take((size_t)M * C * 2);        // 32MB; reused as aggb
  bf16_t* Wqkvb = (bf16_t*)take((size_t)C3 * C * 2);       // 6MB  [3C, C]
  bf16_t* Wob   = (bf16_t*)take((size_t)C * C * 2);        // 2MB
  bf16_t* W1b   = (bf16_t*)take((size_t)Hd * C * 2);       // 8MB
  bf16_t* W2b   = (bf16_t*)take((size_t)C * Hd * 2);       // 8MB
  bf16_t* qkvb  = (bf16_t*)take((size_t)M * C3 * 2);       // 96MB; reused as xA/xC
  char*   big   = (char*)take((size_t)M * Hd * 2);         // 128MB: P8 (8*L*L f32) then yb
  float*  pdiag = (float*)take((size_t)B * NCH * L * 4);   // 1MB partial diag sums
  float*  mc    = (float*)take((size_t)B * L * 4);
  float*  alp   = (float*)take((size_t)B * TK * 4);
  int*    dly   = (int*)take((size_t)B * TK * 4);
  bf16_t* xBb   = (bf16_t*)take((size_t)M * C * 2);        // 32MB
  float*  xB    = (float*)take((size_t)M * C * 4);         // 64MB

  bf16_t* aggb = xnb;            // xn dead after fused qkv GEMM
  float*  P8   = (float*)big;    // 8 * L * L * 4 = 128MB
  bf16_t* yb   = (bf16_t*)big;   // P dead after diag stage
  float*  xA   = (float*)qkvb;   // qkv dead after aggregation (64MB < 96MB)
  float*  xC   = xA;             // xA dead after movavg1

  const dim3 blk(BDIM);

  // 1) LayerNorm -> bf16
  layernorm_bf16<<<M, blk, 0, stream>>>(x, ln_g, ln_b, xnb, C);

  // 2) weights -> bf16 (q/k/v fused into [3C, C])
  cvt_f32_to_bf16<<<(C * C) / BDIM, blk, 0, stream>>>(Wq, Wqkvb,                     C * C);
  cvt_f32_to_bf16<<<(C * C) / BDIM, blk, 0, stream>>>(Wk, Wqkvb + (size_t)C * C,     C * C);
  cvt_f32_to_bf16<<<(C * C) / BDIM, blk, 0, stream>>>(Wv, Wqkvb + (size_t)2 * C * C, C * C);
  cvt_f32_to_bf16<<<(C * C) / BDIM, blk, 0, stream>>>(Wo, Wob, C * C);
  cvt_f32_to_bf16<<<(Hd * C) / BDIM, blk, 0, stream>>>(W1, W1b, Hd * C);
  cvt_f32_to_bf16<<<(C * Hd) / BDIM, blk, 0, stream>>>(W2, W2b, C * Hd);

  // 3) fused qkv = xn * Wqkv^T   -> bf16 [M, 3072]
  dim3 gQKV(C3 / 256, M / 128, 1);
  gemm_wmma_bf16<1><<<gQKV, blk, 0, stream>>>(xnb, C, 0, Wqkvb, C, 0,
                                              nullptr, qkvb, C3, 0,
                                              nullptr, nullptr, 0, C);

  // 4) all 8 batches of P_b = q_b * k_b^T in one launch, then diagonal means
  dim3 gP(L / 256, L / 128, B);
  gemm_wmma_bf16<0><<<gP, blk, 0, stream>>>(qkvb,            C3, (size_t)L * C3,
                                            qkvb + C,        C3, (size_t)L * C3,
                                            P8, nullptr, L, (size_t)L * L,
                                            nullptr, nullptr, 0, C);
  dim3 gDP(NCH, B);
  diag_partial<<<gDP, blk, 0, stream>>>(P8, pdiag, L);
  dim3 gDF(L / BDIM, B);
  diag_finalize<<<gDF, blk, 0, stream>>>(pdiag, mc, L, NCH, 1.0f / (float)C);

  // 5) top-7 lags + softmax, then weighted circular gather of V -> bf16
  topk_softmax<<<B, blk, 0, stream>>>(mc, alp, dly, L, TK);
  dim3 gA((L * C) / BDIM, B);
  aggregate_topk<<<gA, blk, 0, stream>>>(qkvb + 2 * C, C3, alp, dly, aggb, L, C, TK);

  // 6) xA = x + agg*Wo^T + bo
  dim3 gO(C / 256, M / 128, 1);
  gemm_wmma_bf16<2><<<gO, blk, 0, stream>>>(aggb, C, 0, Wob, C, 0,
                                            xA, nullptr, C, 0,
                                            bo, x, C, C);

  // 7) xB = xA - movavg(xA)
  movavg_sub<<<gA, blk, 0, stream>>>(xA, xB, L, C);
  cvt_f32_to_bf16<<<(M * C) / BDIM, blk, 0, stream>>>(xB, xBb, M * C);

  // 8) FFN: yb = GELU(xB * W1^T) bf16 ; xC = xB + yb * W2^T
  dim3 g1(Hd / 256, M / 128, 1);
  gemm_wmma_bf16<3><<<g1, blk, 0, stream>>>(xBb, C, 0, W1b, C, 0,
                                            nullptr, yb, Hd, 0,
                                            nullptr, nullptr, 0, C);
  dim3 g2(C / 256, M / 128, 1);
  gemm_wmma_bf16<4><<<g2, blk, 0, stream>>>(yb, Hd, 0, W2b, Hd, 0,
                                            xC, nullptr, C, 0,
                                            nullptr, xB, C, Hd);

  // 9) out = xC - movavg(xC)
  movavg_sub<<<gA, blk, 0, stream>>>(xC, (float*)d_out, L, C);
}
